// ForwardSelf_22754736734589
// MI455X (gfx1250) — matrix operand order repaired and checked by reading, not compiled, not run on hardware
//
#include <hip/hip_runtime.h>
#include <hip/hip_bf16.h>
#include <math.h>

typedef __attribute__((ext_vector_type(16))) _Float16 v16h;
typedef __attribute__((ext_vector_type(8)))  _Float16 v8h;
typedef __attribute__((ext_vector_type(8)))  float    v8f;
typedef __attribute__((ext_vector_type(4)))  float    v4f;

#define T_STEPS 1000
#define NVIS 256
#define NHID 256
#define NNEU 512            // n_vis + n_hid
#define WLEN 20
#define NSAMP 64
#define SPB 16              // samples per block (WMMA M dimension)
#define NBLK (NSAMP / SPB)  // 4 blocks
#define THREADS 512         // 16 waves (wave32)
#define KTILES (NNEU / 32)  // 16 WMMA K-steps of 32
#define NTILES (NHID / 16)  // 16 N-tiles, one per wave

// ---------------------------------------------------------------------------
// Kernel 0: pack weight (f32 [256][512]) into WMMA B-fragment f16 layout in ws
// and zero the ll outputs.
// B fragment (32x16 f16): lanes 0-15 -> N=lane, K={0..7,16..23}[j]; lanes 16-31 -> N=lane-16,
// K={8..15,24..31}[j].  Flat index (((nt*16 + kt)*32 + lane)*16 + j).
// ---------------------------------------------------------------------------
__global__ void pack_weights(const float* __restrict__ w,
                             _Float16* __restrict__ bpack,
                             float* __restrict__ ll, int ll_n) {
  int i = blockIdx.x * blockDim.x + threadIdx.x;  // 0 .. 131071
  if (i < NTILES * KTILES * 32 * 16) {
    int j    = i & 15;
    int lane = (i >> 4) & 31;
    int kt   = (i >> 9) & 15;
    int nt   = (i >> 13) & 15;
    int n = lane & 15;
    int k = kt * 32 + ((lane < 16) ? 0 : 8) + ((j < 8) ? j : 8 + j);
    int h = nt * 16 + n;
    bpack[i] = (_Float16)w[h * NNEU + k];        // B[k][n] = weight[h][k]
  }
  if (i < ll_n) ll[i] = 0.0f;
}

// A fragment (16x32 f16) from natural row-major inp[s][k]:
// lane l<16 : M=l,    j=0..7 -> K=kt*32+{0..7},  j=8..15 -> K=kt*32+{16..23}
// lane l>=16: M=l-16, j=0..7 -> K=kt*32+{8..15}, j=8..15 -> K=kt*32+{24..31}
// -> two contiguous 16-byte runs: base and base+16 halfs (2x ds_load_b128).
__device__ __forceinline__ v16h load_a(const _Float16* p) {
  v16h a;
  __builtin_memcpy(&a, p, 16);
  __builtin_memcpy(reinterpret_cast<char*>(&a) + 16, p + 16, 16);
  return a;
}

// Workgroup barrier that drains ONLY LDS (DScnt), not global stores.
// CDNA5 split counters + split barrier: output stores (never re-read by this
// kernel) are kept OFF the per-step critical path; s_endpgm's implicit
// wait-idle guarantees their final completion.
__device__ __forceinline__ void wg_barrier_lds_only() {
  asm volatile("s_wait_dscnt 0x0\n\t"
               "s_barrier_signal -1\n\t"
               "s_barrier_wait -1" ::: "memory");
}

// ---------------------------------------------------------------------------
// Main persistent-scan kernel: one block = 16 samples = one WMMA M-tile.
// 16 waves; wave w owns hidden units [16w,16w+16) with its 16 B K-fragments
// resident in VGPRs for all 1000 steps.
// ---------------------------------------------------------------------------
__global__ void __launch_bounds__(THREADS, 1)
glm_scan(const float* __restrict__ vis,
         const float* __restrict__ bias,
         const float* __restrict__ basis,
         const _Float16* __restrict__ bpack,
         float* __restrict__ spikes_out,
         float* __restrict__ conv_out,
         float* __restrict__ ll_out) {
  // CDNA5 320KB/WGP LDS: 160KB f16 spike ring + 16KB f16 staged input.
  __shared__ _Float16 ring[WLEN * SPB * NHID];   // [slot][s][h]
  __shared__ _Float16 inp[SPB * NNEU];           // natural [s][k]
  __shared__ float    lgtab[20];                 // log(k!) for k=0..7(+pad)

  const int tid  = threadIdx.x;
  const int lane = tid & 31;
  const int wave = tid >> 5;        // N-tile id, 0..15
  const int s0   = blockIdx.x * SPB;

  // zero ring (history before t=0; also covers slots unwritten during t<20)
  for (int i = tid; i < (WLEN * SPB * NHID) / 2; i += THREADS)
    reinterpret_cast<int*>(ring)[i] = 0;
  if (tid < 20) lgtab[tid] = lgammaf((float)tid + 1.0f);

  // basis as f16 (conv accumulates in packed f16)
  _Float16 bash[WLEN];
#pragma unroll
  for (int d = 0; d < WLEN; ++d) bash[d] = (_Float16)basis[d];

  // persistent weight B fragments (16 x v16h = 128 VGPRs)
  v16h bf[KTILES];
#pragma unroll
  for (int kt = 0; kt < KTILES; ++kt)
    bf[kt] = *(const v16h*)(bpack + ((size_t)(wave * KTILES + kt) * 32 + lane) * 16);

  // C/D layout: VGPR r -> M = r + (lane<16?0:8), N = lane & 15
  const int   nloc      = lane & 15;
  const int   h_mine    = wave * 16 + nloc;
  const float bias_mine = bias[h_mine];
  const int   mbase     = (lane < 16) ? 0 : 8;

  // persistent per-element RNG state + ll accumulators
  unsigned rngs[8];
  float ll_acc[8];
#pragma unroll
  for (int r = 0; r < 8; ++r) {
    int gs = s0 + mbase + r;
    rngs[r] = (((unsigned)gs * 2654435761u) ^ ((unsigned)h_mine * 40503u) ^
               0x9e3779b9u) | 1u;
    ll_acc[r] = 0.0f;
  }

  // phase-1 partition: fixed (s, 8-wide q run) per thread
  const int sP = tid >> 5;          // 0..15
  const int q0 = (tid & 31) * 8;    // 0..248

  // phase-2 A base: lane's two 16B runs start at offA within each K-tile
  const int sA   = lane & 15;
  const int offA = (lane < 16) ? 0 : 8;
  const _Float16* abase = &inp[sA * NNEU + offA];

  __syncthreads();

  for (int t = 0; t < T_STEPS; ++t) {
    // ring slot for age d (positive mod; unwritten slots are still zero)
    int slots[WLEN];
    {
      int sl = t % WLEN;
#pragma unroll
      for (int d = 1; d <= WLEN; ++d) {
        sl = (sl == 0) ? (WLEN - 1) : (sl - 1);
        slots[d - 1] = sl;
      }
    }

    // ---- phase 1: inp = [vis_t | conv_t], vectorized ----------------------
    {
      // visible drive (k = q0..q0+7), broadcast over samples
      const float* vp = vis + t * NVIS + q0;
      v4f v0 = *(const v4f*)vp;
      v4f v1 = *(const v4f*)(vp + 4);
      v8h vh;
#pragma unroll
      for (int i = 0; i < 4; ++i) {
        vh[i]     = (_Float16)v0[i];
        vh[i + 4] = (_Float16)v1[i];
      }
      *(v8h*)&inp[sP * NNEU + q0] = vh;                 // ds_store_b128

      // FIR conv over 20 lags, two independent v_pk_fma_f16 chains
      v8h ca = {0, 0, 0, 0, 0, 0, 0, 0};
      v8h cb = {0, 0, 0, 0, 0, 0, 0, 0};
#pragma unroll
      for (int d = 1; d <= WLEN; d += 2) {
        const v8h r0 = *(const v8h*)&ring[slots[d - 1] * (SPB * NHID) +
                                          sP * NHID + q0];
        const v8h r1 = *(const v8h*)&ring[slots[d] * (SPB * NHID) +
                                          sP * NHID + q0];
        const _Float16 b0 = bash[d - 1], b1 = bash[d];
        const v8h bs0 = {b0, b0, b0, b0, b0, b0, b0, b0};
        const v8h bs1 = {b1, b1, b1, b1, b1, b1, b1, b1};
        ca += r0 * bs0;
        cb += r1 * bs1;
      }
      const v8h cacc = ca + cb;
      *(v8h*)&inp[sP * NNEU + NVIS + q0] = cacc;        // ds_store_b128

      v4f c0, c1;
#pragma unroll
      for (int i = 0; i < 4; ++i) {
        c0[i] = (float)cacc[i];
        c1[i] = (float)cacc[i + 4];
      }
      float* cp = conv_out + ((size_t)(s0 + sP) * T_STEPS + t) * NHID + q0;
      *(v4f*)cp       = c0;                             // global_store_b128
      *(v4f*)(cp + 4) = c1;
    }
    wg_barrier_lds_only();

    // ---- phase 2: WMMA GEMM, dual acc chains + double-buffered A ----------
    v8f acc0, acc1;
#pragma unroll
    for (int r = 0; r < 8; ++r) { acc0[r] = bias_mine; acc1[r] = 0.0f; }

    v16h a0 = load_a(abase);
    v16h a1 = load_a(abase + 32);
#pragma unroll
    for (int kt = 0; kt < KTILES; kt += 2) {
      v16h an0 = a0, an1 = a1;
      if (kt + 2 < KTILES) an0 = load_a(abase + (kt + 2) * 32);
      if (kt + 3 < KTILES) an1 = load_a(abase + (kt + 3) * 32);
      acc0 = __builtin_amdgcn_wmma_f32_16x16x32_f16(false, a0, false, bf[kt],
                                                    (short)0, acc0, false, false);
      acc1 = __builtin_amdgcn_wmma_f32_16x16x32_f16(false, a1, false, bf[kt + 1],
                                                    (short)0, acc1, false, false);
      a0 = an0; a1 = an1;
    }

    // ---- phase 3: sigmoid -> Poisson (branch-free inverse CDF) ------------
    const int wbase = (t % WLEN) * (SPB * NHID);
#pragma unroll
    for (int r = 0; r < 8; ++r) {
      const int m  = mbase + r;
      const int gs = s0 + m;
      float rate = 1.0f / (1.0f + __expf(-(acc0[r] + acc1[r])));
      rate = fminf(fmaxf(rate, 1e-7f), 1.0f - 1e-7f);
      // one uniform
      unsigned rng = rngs[r];
      rng ^= rng << 13; rng ^= rng >> 17; rng ^= rng << 5;
      rngs[r] = rng;
      const float u = (float)(rng >> 8) * (1.0f / 16777216.0f);
      // truncated inverse-CDF Poisson: lambda <= 1 -> P(k>6) < 1e-4
      float term = __expf(-rate);   // P(0)
      float cdf  = term;
      int k = (u > cdf);
      term *= rate;               cdf += term; k += (u > cdf);
      term *= rate * 0.5f;        cdf += term; k += (u > cdf);
      term *= rate * (1.f / 3.f); cdf += term; k += (u > cdf);
      term *= rate * 0.25f;       cdf += term; k += (u > cdf);
      term *= rate * 0.2f;        cdf += term; k += (u > cdf);
      const float sp = (float)k;
      spikes_out[((size_t)gs * T_STEPS + t) * NHID + h_mine] = sp;
      ring[wbase + m * NHID + h_mine] = (_Float16)sp;
      ll_acc[r] += sp * __logf(rate) - rate - lgtab[k];
    }
    if (t + 1 < T_STEPS) __builtin_prefetch(vis + (t + 1) * NVIS, 0, 1);
    wg_barrier_lds_only();
  }

#pragma unroll
  for (int r = 0; r < 8; ++r)
    atomicAdd(&ll_out[s0 + mbase + r], ll_acc[r]);
}

// ---------------------------------------------------------------------------
extern "C" void kernel_launch(void* const* d_in, const int* in_sizes, int n_in,
                              void* d_out, int out_size, void* d_ws, size_t ws_size,
                              hipStream_t stream) {
  const float* vis    = (const float*)d_in[0];  // (1000, 256) f32
  const float* weight = (const float*)d_in[1];  // (256, 512) f32
  const float* bias   = (const float*)d_in[2];  // (256,) f32
  const float* basis  = (const float*)d_in[3];  // (20,) f32
  // d_in[4] = n_samples (64), matched by NSAMP

  float* out        = (float*)d_out;
  float* spikes_out = out;                                        // (64,1000,256)
  float* conv_out   = out + (size_t)NSAMP * T_STEPS * NHID;       // (64,1000,256)
  float* ll_out     = out + (size_t)2 * NSAMP * T_STEPS * NHID;   // (64,)

  _Float16* bpack = (_Float16*)d_ws;  // 256KB packed f16 WMMA-B fragments

  pack_weights<<<512, 256, 0, stream>>>(weight, bpack, ll_out, NSAMP);
  glm_scan<<<NBLK, THREADS, 0, stream>>>(vis, bias, basis, bpack,
                                         spikes_out, conv_out, ll_out);
}
